// MixedSparseSingleLayer_14559939134114
// MI455X (gfx1250) — matrix operand order repaired and checked by reading, not compiled, not run on hardware
//
#include <hip/hip_runtime.h>
#include <hip/hip_bf16.h>

// ---------------------------------------------------------------------------
// MI455X (gfx1250) transformer layer: RMSNorm -> LoRA-QKV -> RoPE -> causal
// flash attention -> LoRA-O + residual -> RMSNorm -> LoRA-up + SiLU ->
// LoRA-down + residual.
// Big GEMMs + attention: V_WMMA_F32_16X16X32_BF16.
// GEMM tile staging: Tensor Data Mover (tensor_load_to_lds) double-buffered.
// ---------------------------------------------------------------------------

#define TB 2
#define TS 2048
#define TD 2048
#define TH 16
#define THD 128
#define TF 8192
#define TR 16
#define TM (TB * TS)   // 4096 rows

typedef unsigned int  uint;
typedef unsigned short ushort;
typedef __attribute__((ext_vector_type(16))) __bf16 v16bf;
typedef __attribute__((ext_vector_type(8)))  float  v8f;
typedef __attribute__((ext_vector_type(4)))  unsigned int u32x4;
typedef __attribute__((ext_vector_type(8)))  int i32x8;
typedef __attribute__((ext_vector_type(4)))  int i32x4;

static __device__ inline ushort f2bf(float f) {
    uint u = __float_as_uint(f);
    u += 0x7FFFu + ((u >> 16) & 1u);           // round to nearest even
    return (ushort)(u >> 16);
}
static __device__ inline __bf16 bf_from_bits(ushort u) {
    union { ushort s; __bf16 b; } c; c.s = u; return c.b;
}

// ---------------------------------------------------------------------------
// TDM: 2D bf16 tile (rows x 32) from global (row stride = K elems) into LDS.
// D# per CDNA5 ISA ch.8: group0 {count, lds_addr, global_addr, type=2},
// group1 {data_size=1(2B), tensor_dim0=K, tensor_dim1=rows, tile=32 x rows,
//         tensor_dim0_stride=K}.  groups 2/3 zero (2D tensor).
// Toolchain uses the 6-arg builtin: (g0, g1, g2, g3, extra, cpol).
// ---------------------------------------------------------------------------
static __device__ inline void tdm_load_tile(unsigned ldsOff, const ushort* gptr,
                                            int rows, int K) {
    unsigned long long ga = (unsigned long long)gptr;
    u32x4 g0;
    g0[0] = 1u;                                           // count=1 (valid)
    g0[1] = ldsOff;                                       // lds_addr (bytes)
    g0[2] = (unsigned)(ga & 0xFFFFFFFFu);                 // global_addr lo
    g0[3] = (unsigned)((ga >> 32) & 0x01FFFFFFu) | (2u << 30);  // addr hi | type=2
    unsigned td0 = (unsigned)K;                           // tensor_dim0
    unsigned td1 = (unsigned)rows;                        // tensor_dim1
    unsigned long long st0 = (unsigned long long)K;       // tensor_dim0_stride
    i32x8 g1;
    g1[0] = (int)(1u << 16);                              // data_size=1 (2 bytes)
    g1[1] = (int)((td0 & 0xFFFFu) << 16);                 // tensor_dim0[15:0]
    g1[2] = (int)(((td0 >> 16) & 0xFFFFu) | ((td1 & 0xFFFFu) << 16));
    g1[3] = (int)(((td1 >> 16) & 0xFFFFu) | (32u << 16)); // tile_dim0 = 32
    g1[4] = (int)((unsigned)rows & 0xFFFFu);              // tile_dim1, tile_dim2=0
    g1[5] = (int)(unsigned)(st0 & 0xFFFFFFFFu);           // stride lo
    g1[6] = (int)(unsigned)((st0 >> 32) & 0xFFFFu);       // stride hi, dim1_stride=0
    g1[7] = 0;
    i32x4 gz4 = {0, 0, 0, 0};
    i32x8 gz8 = {0, 0, 0, 0, 0, 0, 0, 0};
    __builtin_amdgcn_tensor_load_to_lds(g0, g1, gz4, gz4, gz8, 0);
}

// ---------------------------------------------------------------------------
// Weight convert + transpose: Wt[n][k] = bf16(W[k][n])
// ---------------------------------------------------------------------------
__global__ void wcvt_t_kernel(const float* __restrict__ w, ushort* __restrict__ wt,
                              int K, int N) {
    long i = (long)blockIdx.x * blockDim.x + threadIdx.x;
    long tot = (long)K * N;
    if (i >= tot) return;
    int n = (int)(i / K);
    int k = (int)(i % K);
    wt[i] = f2bf(w[(long)k * N + n]);
}

// ---------------------------------------------------------------------------
// RMSNorm: one block per row; writes fp32 and bf16 copies
// ---------------------------------------------------------------------------
__global__ __launch_bounds__(256)
void rmsnorm_kernel(const float* __restrict__ x, const float* __restrict__ w,
                    float* __restrict__ yf, ushort* __restrict__ yb, int Dm) {
    int row = blockIdx.x;
    const float* xr = x + (long)row * Dm;
    float ss = 0.f;
    for (int i = threadIdx.x; i < Dm; i += blockDim.x) { float v = xr[i]; ss += v * v; }
    __shared__ float sm[256];
    sm[threadIdx.x] = ss; __syncthreads();
    for (int s = 128; s > 0; s >>= 1) {
        if (threadIdx.x < s) sm[threadIdx.x] += sm[threadIdx.x + s];
        __syncthreads();
    }
    float scale = rsqrtf(sm[0] / (float)Dm + 1e-10f);
    for (int i = threadIdx.x; i < Dm; i += blockDim.x) {
        float v = xr[i] * scale * w[i];
        yf[(long)row * Dm + i] = v;
        yb[(long)row * Dm + i] = f2bf(v);
    }
}

// ---------------------------------------------------------------------------
// LoRA intermediate: t[M][16] = A[M][K] @ la[K][16]   (tiny; 0.1% of FLOPs)
// ---------------------------------------------------------------------------
__global__ __launch_bounds__(256)
void lora_t_kernel(const float* __restrict__ a, const float* __restrict__ la,
                   float* __restrict__ t, int K) {
    int row = blockIdx.x;
    float acc[TR];
#pragma unroll
    for (int r = 0; r < TR; r++) acc[r] = 0.f;
    for (int k = threadIdx.x; k < K; k += blockDim.x) {
        float av = a[(long)row * K + k];
        const float* lr = la + (long)k * TR;
#pragma unroll
        for (int r = 0; r < TR; r++) acc[r] += av * lr[r];
    }
    __shared__ float sm[256];
    for (int r = 0; r < TR; r++) {
        sm[threadIdx.x] = acc[r]; __syncthreads();
        for (int s = 128; s > 0; s >>= 1) {
            if (threadIdx.x < s) sm[threadIdx.x] += sm[threadIdx.x + s];
            __syncthreads();
        }
        if (threadIdx.x == 0) t[(long)row * TR + r] = sm[0];
        __syncthreads();
    }
}

// ---------------------------------------------------------------------------
// Tiled WMMA GEMM:  out[M][N] = A_bf16[M][K] @ Wt_bf16[N][K]^T
//   Block 256 thr = 8 waves as 2(M) x 4(N); wave tile 32x32 (4 WMMA / K-step).
//   Block tile 64x128, K step 32.  Tiles staged by TDM (wave 0 issues
//   tensor_load_to_lds), double-buffered LDS, s_wait_tensorcnt pipelining.
//   Epilogue: + bias[n] + t[m][:]@lb[:][n] (+ residual) (SiLU opt) (+bf16 out)
// ---------------------------------------------------------------------------
__global__ __launch_bounds__(256)
void gemm_wmma_kernel(const ushort* __restrict__ A, const ushort* __restrict__ Wt,
                      const float* __restrict__ bias,
                      const float* __restrict__ tl, const float* __restrict__ lb,
                      const float* __restrict__ res,
                      float* __restrict__ outF, ushort* __restrict__ outB,
                      int N, int K, int applySilu) {
    __shared__ __align__(16) ushort lA[2][64 * 32];    // [buf][m][k]
    __shared__ __align__(16) ushort lB[2][128 * 32];   // [buf][n][k]
    const int tid  = threadIdx.x;
    const int lane = tid & 31;
    const int wave = tid >> 5;
    const int wm = wave & 1;          // 2 waves along M (32 rows each)
    const int wn = wave >> 1;         // 4 waves along N (32 cols each)
    const int half = lane >> 4;
    const int l15  = lane & 15;
    const int mBlock = blockIdx.y * 64;
    const int nBlock = blockIdx.x * 128;

    const ushort* Abase = A  + (long)mBlock * K;
    const ushort* Bbase = Wt + (long)nBlock * K;
    const int nK = K / 32;

    v8f c00 = {}, c01 = {}, c10 = {}, c11 = {};

    // prime buffer 0 via TDM
    if (wave == 0) {
        tdm_load_tile((unsigned)(unsigned long long)&lA[0][0], Abase, 64, K);
        tdm_load_tile((unsigned)(unsigned long long)&lB[0][0], Bbase, 128, K);
    }

    for (int kt = 0; kt < nK; kt++) {
        const int cur = kt & 1;
        if (wave == 0) {
            if (kt + 1 < nK) {
                const int nxt = (kt + 1) & 1;
                tdm_load_tile((unsigned)(unsigned long long)&lA[nxt][0],
                              Abase + (kt + 1) * 32, 64, K);
                tdm_load_tile((unsigned)(unsigned long long)&lB[nxt][0],
                              Bbase + (kt + 1) * 32, 128, K);
                __builtin_amdgcn_s_wait_tensorcnt(2);   // current pair done
            } else {
                __builtin_amdgcn_s_wait_tensorcnt(0);
            }
        }
        __syncthreads();   // tile `cur` visible to all waves

        // A fragments: k(i,l) = half*16 + i
        v16bf a0, a1;
        {
            const ushort* p = &lA[cur][(wm * 32 + l15) * 32 + half * 16];
            *(uint4*)&a0         = *(const uint4*)p;
            *(((uint4*)&a0) + 1) = *(const uint4*)(p + 8);
            const ushort* q = p + 16 * 32;
            *(uint4*)&a1         = *(const uint4*)q;
            *(((uint4*)&a1) + 1) = *(const uint4*)(q + 8);
        }
        // B fragments: element i -> k = half*16 + i, n = l15
        v16bf b0, b1;
        {
            const ushort* p = &lB[cur][(wn * 32 + l15) * 32 + half * 16];
            *(uint4*)&b0         = *(const uint4*)p;
            *(((uint4*)&b0) + 1) = *(const uint4*)(p + 8);
            const ushort* q = p + 16 * 32;
            *(uint4*)&b1         = *(const uint4*)q;
            *(((uint4*)&b1) + 1) = *(const uint4*)(q + 8);
        }
        c00 = __builtin_amdgcn_wmma_f32_16x16x32_bf16(false, a0, false, b0,
                                                      (short)0, c00, false, false);
        c01 = __builtin_amdgcn_wmma_f32_16x16x32_bf16(false, a0, false, b1,
                                                      (short)0, c01, false, false);
        c10 = __builtin_amdgcn_wmma_f32_16x16x32_bf16(false, a1, false, b0,
                                                      (short)0, c10, false, false);
        c11 = __builtin_amdgcn_wmma_f32_16x16x32_bf16(false, a1, false, b1,
                                                      (short)0, c11, false, false);
        __syncthreads();   // done reading tile `cur` before TDM overwrites it
    }

    // Epilogue: D-layout element v -> m = v + 8*half, n = l15
#pragma unroll
    for (int fm = 0; fm < 2; fm++) {
#pragma unroll
        for (int fn = 0; fn < 2; fn++) {
            v8f c = fm ? (fn ? c11 : c10) : (fn ? c01 : c00);
            int n = nBlock + wn * 32 + fn * 16 + l15;
            float bn = bias[n];
#pragma unroll
            for (int v = 0; v < 8; v++) {
                int m = mBlock + wm * 32 + fm * 16 + half * 8 + v;
                float acc = c[v] + bn;
                const float* tr = tl + (long)m * TR;
#pragma unroll
                for (int r = 0; r < TR; r++) acc += tr[r] * lb[(long)r * N + n];
                if (res) acc += res[(long)m * N + n];
                if (applySilu) acc = acc / (1.0f + __expf(-acc));   // x*sigmoid(x)
                outF[(long)m * N + n] = acc;
                if (outB) outB[(long)m * N + n] = f2bf(acc);
            }
        }
    }
}

// ---------------------------------------------------------------------------
// RoPE + head-layout prep:
//   qh,kh : bf16 [B,H,S,HD] with rotary applied
//   vt    : bf16 [B,H,HD,S] (transposed so P@V B-fragments are contiguous)
// ---------------------------------------------------------------------------
__global__ __launch_bounds__(256)
void prep_qkv_kernel(const float* __restrict__ q, const float* __restrict__ k,
                     const float* __restrict__ v,
                     const float* __restrict__ cosp, const float* __restrict__ sinp,
                     ushort* __restrict__ qh, ushort* __restrict__ kh,
                     ushort* __restrict__ vt) {
    long bs = blockIdx.x;               // 0 .. B*S-1
    int b = (int)(bs / TS), s = (int)(bs % TS);
    const float* qr = q + bs * TD;
    const float* kr = k + bs * TD;
    const float* vr = v + bs * TD;
    for (int d = threadIdx.x; d < TD; d += blockDim.x) {
        int h  = d >> 7;
        int hd = d & 127;
        float cv = cosp[s * THD + hd];
        float sv = sinp[s * THD + hd];
        float qrot = (hd < 64) ? -qr[d + 64] : qr[d - 64];
        float krot = (hd < 64) ? -kr[d + 64] : kr[d - 64];
        long hidx = ((((long)b * TH + h) * TS) + s) * THD + hd;
        qh[hidx] = f2bf(qr[d] * cv + qrot * sv);
        kh[hidx] = f2bf(kr[d] * cv + krot * sv);
        vt[(((long)b * TH + h) * THD + hd) * TS + s] = f2bf(vr[d]);
    }
}

// ---------------------------------------------------------------------------
// Flash attention (causal), one wave per 16 query rows.
//   S-tile 16x32 via WMMA (K-dim=HD in 4 slices), online softmax in registers
//   (C/D layout: element v <-> row v + 8*half), P->A-fragment via LDS
//   round-trip, O accumulated as 8 WMMA C fragments (HD=128).
// ---------------------------------------------------------------------------
__global__ __launch_bounds__(32)
void flash_attn_kernel(const ushort* __restrict__ qh, const ushort* __restrict__ kh,
                       const ushort* __restrict__ vt,
                       float* __restrict__ oF, ushort* __restrict__ oB) {
    const int qt   = blockIdx.x;        // query tile (16 rows)
    const int bh   = blockIdx.y;        // b*H + h
    const int b    = bh / TH;
    const int h    = bh % TH;
    const int lane = threadIdx.x;
    const int half = lane >> 4;
    const int l15  = lane & 15;

    const ushort* Qp = qh + (long)bh * TS * THD;
    const ushort* Kp = kh + (long)bh * TS * THD;
    const ushort* Vp = vt + (long)bh * THD * TS;

    // Preload Q fragments (4 slices of 32 along HD)
    v16bf qf[4];
    const int qm = qt * 16 + l15;
#pragma unroll
    for (int ds = 0; ds < 4; ds++) {
        const ushort* p = Qp + (long)qm * THD + ds * 32 + half * 16;
        *(uint4*)&qf[ds]         = *(const uint4*)p;
        *(((uint4*)&qf[ds]) + 1) = *(const uint4*)(p + 8);
    }

    v8f o[8] = {};
    float mst[8], lst[8];
#pragma unroll
    for (int v = 0; v < 8; v++) { mst[v] = -1e30f; lst[v] = 0.f; }

    __shared__ __align__(16) float pLDS[16 * 32];
    const float scale = 0.08838834764831845f;        // 1/sqrt(128)
    const int ktMax = (qt * 16 + 16 + 31) / 32;      // causal tile bound

    for (int kt = 0; kt < ktMax; kt++) {
        const int kbase = kt * 32;
        v8f s0 = {}, s1 = {};
#pragma unroll
        for (int ds = 0; ds < 4; ds++) {
            v16bf b0, b1;
            const ushort* p0 = Kp + (long)(kbase + l15) * THD + ds * 32 + half * 16;
            *(uint4*)&b0         = *(const uint4*)p0;
            *(((uint4*)&b0) + 1) = *(const uint4*)(p0 + 8);
            const ushort* p1 = Kp + (long)(kbase + 16 + l15) * THD + ds * 32 + half * 16;
            *(uint4*)&b1         = *(const uint4*)p1;
            *(((uint4*)&b1) + 1) = *(const uint4*)(p1 + 8);
            s0 = __builtin_amdgcn_wmma_f32_16x16x32_bf16(false, qf[ds], false, b0,
                                                         (short)0, s0, false, false);
            s1 = __builtin_amdgcn_wmma_f32_16x16x32_bf16(false, qf[ds], false, b1,
                                                         (short)0, s1, false, false);
        }

        // scale + causal mask + online softmax
        float fscale[8];
#pragma unroll
        for (int v = 0; v < 8; v++) {
            int qg  = qt * 16 + half * 8 + v;
            int k0g = kbase + l15;
            int k1g = kbase + 16 + l15;
            s0[v] = (k0g <= qg) ? s0[v] * scale : -1e30f;
            s1[v] = (k1g <= qg) ? s1[v] * scale : -1e30f;

            float mx = fmaxf(s0[v], s1[v]);
#pragma unroll
            for (int off = 8; off >= 1; off >>= 1)
                mx = fmaxf(mx, __shfl_xor(mx, off, 32));
            float mnew = fmaxf(mst[v], mx);
            float e0 = __expf(s0[v] - mnew);
            float e1 = __expf(s1[v] - mnew);
            float sum = e0 + e1;
#pragma unroll
            for (int off = 8; off >= 1; off >>= 1)
                sum += __shfl_xor(sum, off, 32);
            float fz = __expf(mst[v] - mnew);
            lst[v] = lst[v] * fz + sum;
            mst[v] = mnew;
            fscale[v] = fz;
            s0[v] = e0; s1[v] = e1;
        }

        // P (fp32, D-layout) -> LDS -> bf16 A-fragment layout
#pragma unroll
        for (int v = 0; v < 8; v++) {
            pLDS[(half * 8 + v) * 32 + l15]      = s0[v];
            pLDS[(half * 8 + v) * 32 + 16 + l15] = s1[v];
        }
        __syncthreads();
        v16bf pa;
#pragma unroll
        for (int i = 0; i < 16; i++) {
            int kk = half * 16 + i;
            pa[i] = bf_from_bits(f2bf(pLDS[l15 * 32 + kk]));
        }
        __syncthreads();

        // O = diag(fscale)*O + P @ V   (8 chunks of 16 along HD)
#pragma unroll
        for (int hc = 0; hc < 8; hc++) {
#pragma unroll
            for (int v = 0; v < 8; v++) o[hc][v] *= fscale[v];
            v16bf bv;
            const ushort* pv = Vp + (long)(hc * 16 + l15) * TS + kbase + half * 16;
            *(uint4*)&bv         = *(const uint4*)pv;
            *(((uint4*)&bv) + 1) = *(const uint4*)(pv + 8);
            o[hc] = __builtin_amdgcn_wmma_f32_16x16x32_bf16(false, pa, false, bv,
                                                            (short)0, o[hc], false, false);
        }
    }

    // normalize + write back to hidden layout [B,S,D]
#pragma unroll
    for (int hc = 0; hc < 8; hc++) {
#pragma unroll
        for (int v = 0; v < 8; v++) {
            int qg = qt * 16 + half * 8 + v;
            float val = o[hc][v] / lst[v];
            long idx = (((long)b * TS) + qg) * TD + h * THD + hc * 16 + l15;
            oF[idx] = val;
            oB[idx] = f2bf(val);
        }
    }
}

// ---------------------------------------------------------------------------
// Host-side orchestration
// ---------------------------------------------------------------------------
extern "C" void kernel_launch(void* const* d_in, const int* in_sizes, int n_in,
                              void* d_out, int out_size, void* d_ws, size_t ws_size,
                              hipStream_t stream) {
    (void)in_sizes; (void)n_in; (void)out_size; (void)ws_size;

    const float* x    = (const float*)d_in[0];
    const float* nw1  = (const float*)d_in[1];
    const float* nw2  = (const float*)d_in[2];
    const float* cosp = (const float*)d_in[3];
    const float* sinp = (const float*)d_in[4];
    // d_in[5] = attention_mask (pure causal; implemented in-kernel)
    const float* w_q  = (const float*)d_in[6];
    const float* b_q  = (const float*)d_in[7];
    const float* laq  = (const float*)d_in[8];
    const float* lbq  = (const float*)d_in[9];
    const float* w_k  = (const float*)d_in[10];
    const float* b_k  = (const float*)d_in[11];
    const float* lak  = (const float*)d_in[12];
    const float* lbk  = (const float*)d_in[13];
    const float* w_v  = (const float*)d_in[14];
    const float* b_v  = (const float*)d_in[15];
    const float* lav  = (const float*)d_in[16];
    const float* lbv  = (const float*)d_in[17];
    const float* w_o  = (const float*)d_in[18];
    const float* b_o  = (const float*)d_in[19];
    const float* lao  = (const float*)d_in[20];
    const float* lbo  = (const float*)d_in[21];
    const float* w_up = (const float*)d_in[22];
    const float* b_up = (const float*)d_in[23];
    const float* laup = (const float*)d_in[24];
    const float* lbup = (const float*)d_in[25];
    const float* w_dn = (const float*)d_in[26];
    const float* b_dn = (const float*)d_in[27];
    const float* ladn = (const float*)d_in[28];
    const float* lbdn = (const float*)d_in[29];

    char* wsb = (char*)d_ws;
    size_t off = 0;
    auto take = [&](size_t bytes) -> void* {
        off = (off + 255) & ~(size_t)255;
        void* p = wsb + off;
        off += bytes;
        return p;
    };

    ushort* wq_t = (ushort*)take((size_t)TD * TD * 2);
    ushort* wk_t = (ushort*)take((size_t)TD * TD * 2);
    ushort* wv_t = (ushort*)take((size_t)TD * TD * 2);
    ushort* wo_t = (ushort*)take((size_t)TD * TD * 2);
    ushort* wup_t = (ushort*)take((size_t)TD * TF * 2);
    ushort* wdn_t = (ushort*)take((size_t)TF * TD * 2);

    float*  xn1_f = (float*) take((size_t)TM * TD * 4);
    ushort* xn1_b = (ushort*)take((size_t)TM * TD * 2);
    float*  q_f   = (float*) take((size_t)TM * TD * 4);
    float*  k_f   = (float*) take((size_t)TM * TD * 4);
    float*  v_f   = (float*) take((size_t)TM * TD * 4);
    ushort* qh_b  = (ushort*)take((size_t)TM * TD * 2);
    ushort* kh_b  = (ushort*)take((size_t)TM * TD * 2);
    ushort* vt_b  = (ushort*)take((size_t)TM * TD * 2);
    float*  o_f   = (float*) take((size_t)TM * TD * 4);
    ushort* o_b   = (ushort*)take((size_t)TM * TD * 2);
    float*  x1_f  = (float*) take((size_t)TM * TD * 4);
    float*  xn2_f = (float*) take((size_t)TM * TD * 4);
    ushort* xn2_b = (ushort*)take((size_t)TM * TD * 2);
    float*  fn_f  = (float*) take((size_t)TM * TF * 4);
    ushort* fn_b  = (ushort*)take((size_t)TM * TF * 2);
    float*  t_buf = (float*) take((size_t)TM * TR * 4);

    const long ndd = (long)TD * TD;
    const long ndf = (long)TD * TF;

    // 1) weights -> bf16, transposed [N][K]
    wcvt_t_kernel<<<(uint)((ndd + 255) / 256), 256, 0, stream>>>(w_q, wq_t, TD, TD);
    wcvt_t_kernel<<<(uint)((ndd + 255) / 256), 256, 0, stream>>>(w_k, wk_t, TD, TD);
    wcvt_t_kernel<<<(uint)((ndd + 255) / 256), 256, 0, stream>>>(w_v, wv_t, TD, TD);
    wcvt_t_kernel<<<(uint)((ndd + 255) / 256), 256, 0, stream>>>(w_o, wo_t, TD, TD);
    wcvt_t_kernel<<<(uint)((ndf + 255) / 256), 256, 0, stream>>>(w_up, wup_t, TD, TF);
    wcvt_t_kernel<<<(uint)((ndf + 255) / 256), 256, 0, stream>>>(w_dn, wdn_t, TF, TD);

    // 2) RMSNorm #1
    rmsnorm_kernel<<<TM, 256, 0, stream>>>(x, nw1, xn1_f, xn1_b, TD);

    dim3 gemmDD(TD / 128, TM / 64);
    dim3 gemmDF(TF / 128, TM / 64);

    // 3) q/k/v projections with fused LoRA + bias
    lora_t_kernel<<<TM, 256, 0, stream>>>(xn1_f, laq, t_buf, TD);
    gemm_wmma_kernel<<<gemmDD, 256, 0, stream>>>(xn1_b, wq_t, b_q, t_buf, lbq,
                                                 nullptr, q_f, nullptr, TD, TD, 0);
    lora_t_kernel<<<TM, 256, 0, stream>>>(xn1_f, lak, t_buf, TD);
    gemm_wmma_kernel<<<gemmDD, 256, 0, stream>>>(xn1_b, wk_t, b_k, t_buf, lbk,
                                                 nullptr, k_f, nullptr, TD, TD, 0);
    lora_t_kernel<<<TM, 256, 0, stream>>>(xn1_f, lav, t_buf, TD);
    gemm_wmma_kernel<<<gemmDD, 256, 0, stream>>>(xn1_b, wv_t, b_v, t_buf, lbv,
                                                 nullptr, v_f, nullptr, TD, TD, 0);

    // 4) RoPE + head layouts (V transposed)
    prep_qkv_kernel<<<TM, 256, 0, stream>>>(q_f, k_f, v_f, cosp, sinp,
                                            qh_b, kh_b, vt_b);

    // 5) causal flash attention
    dim3 fgrid(TS / 16, TB * TH);
    flash_attn_kernel<<<fgrid, 32, 0, stream>>>(qh_b, kh_b, vt_b, o_f, o_b);

    // 6) output projection + residual -> x1
    lora_t_kernel<<<TM, 256, 0, stream>>>(o_f, lao, t_buf, TD);
    gemm_wmma_kernel<<<gemmDD, 256, 0, stream>>>(o_b, wo_t, b_o, t_buf, lbo,
                                                 x, x1_f, nullptr, TD, TD, 0);

    // 7) RMSNorm #2
    rmsnorm_kernel<<<TM, 256, 0, stream>>>(x1_f, nw2, xn2_f, xn2_b, TD);

    // 8) up projection + SiLU
    lora_t_kernel<<<TM, 256, 0, stream>>>(xn2_f, laup, t_buf, TD);
    gemm_wmma_kernel<<<gemmDF, 256, 0, stream>>>(xn2_b, wup_t, b_up, t_buf, lbup,
                                                 nullptr, fn_f, fn_b, TF, TD, 1);

    // 9) down projection + residual -> final output
    lora_t_kernel<<<TM, 256, 0, stream>>>(fn_f, ladn, t_buf, TF);
    gemm_wmma_kernel<<<gemmDD, 256, 0, stream>>>(fn_b, wdn_t, b_dn, t_buf, lbdn,
                                                 x1_f, (float*)d_out, nullptr, TD, TF, 0);
}
